// VideoMamba3DExtra_35991825940959
// MI455X (gfx1250) — compile-verified, matrix operand order checked by reading, not run on hardware
//
#include <hip/hip_runtime.h>
#include <hip/hip_bf16.h>
#include <string.h>

// ---------------------------------------------------------------------------
// VideoMamba3D extra-block, fused per-fiber bidirectional Mamba layer.
// One workgroup (256 threads = 8 wave32) owns one fiber (T<=32 tokens, D=384).
// All activations live in LDS (319,488 B < 320 KB/WGP). GEMMs use
// v_wmma_f32_16x16x32_bf16 with A fragments from LDS (ds_load_b128) and B
// fragments from bf16 weights pre-packed in fragment order (global_load_b128,
// zero conversion VALU in the inner loop). Selective scan in f32 VALU with
// per-thread register state (3 channels x 16 states).
// ---------------------------------------------------------------------------

#define DM   384     // d_model
#define DI   768     // d_inner
#define TMAX 32      // max sequence length (L axis); H/W axes use T=28
#define DTR  24      // dt_rank
#define DST  16      // d_state
#define NTHREADS 256

typedef __attribute__((ext_vector_type(16))) __bf16 v16bf;
typedef __attribute__((ext_vector_type(8)))  float  v8f;

struct LayerP {
  // small f32 vectors (read directly, L2-resident)
  const float* norm_w;  // [384]
  const float* conv_w;  // [4 x 768]
  const float* conv_b;  // [768]
  const float* b_dt;    // [768]
  const float* A_log;   // [768 x 16]
  const float* D;       // [768]
  // bf16 weights packed in WMMA-B fragment order (in d_ws)
  const unsigned short* pW_in;   // K=384  -> 12 kblk, 96 ntiles
  const unsigned short* pW_x;    // K=768  -> 24 kblk,  4 ntiles (56 -> 64 pad)
  const unsigned short* pW_dt;   // K=32   ->  1 kblk, 48 ntiles (24 -> 32 pad)
  const unsigned short* pW_out;  // K=768  -> 24 kblk, 24 ntiles
};

// native hardware f32 -> bf16 convert (RNE), instead of emulated rounding
__device__ __forceinline__ unsigned short f2bf(float f) {
  __bf16 h = (__bf16)f;
  return __builtin_bit_cast(unsigned short, h);
}
// exact bf16 -> f32 is a single shift
__device__ __forceinline__ float bf2f(unsigned short h) {
  union { unsigned int u; float f; } x; x.u = ((unsigned int)h) << 16;
  return x.f;
}

// A fragment, 16x32 bf16, from LDS. Order used here (same as pack_b_kernel): lane&15 = M, lane>>4 selects
// K-half, element j -> K = 16*(lane>>4) + j.  -> two ds_load_b128 per lane.
__device__ __forceinline__ v16bf load_a_lds(const unsigned short* A, int lda,
                                            int mbase, int kbase, int lane) {
  int m  = mbase + (lane & 15);
  int kb = kbase + ((lane >> 4) << 4);
  union { v16bf v; unsigned short u[16]; } r;
  const unsigned short* p = A + m * lda + kb;
#pragma unroll
  for (int j = 0; j < 16; ++j) r.u[j] = p[j];
  return r.v;
}

// B fragment from packed bf16 weights: lane's 16 values are contiguous
// (32 B) -> two global_load_b128, no conversion ALU.
__device__ __forceinline__ v16bf load_b_pk(const unsigned short* __restrict__ Bp,
                                           int ntiles, int kk, int nt, int lane) {
  const unsigned short* p = Bp + ((((size_t)kk * ntiles + nt) * 32 + lane) << 4);
  union { v16bf v; unsigned short u[16]; } r;
#pragma unroll
  for (int j = 0; j < 16; ++j) r.u[j] = p[j];
  return r.v;
}

__device__ __forceinline__ v8f wmma_bf16(v16bf a, v16bf b, v8f c) {
  return __builtin_amdgcn_wmma_f32_16x16x32_bf16(false, a, false, b,
                                                 (short)0, c, false, false);
}

__device__ __forceinline__ float siluf(float v) {
  return v * (1.f / (1.f + __expf(-v)));
}

// One Mamba layer, in place on the LDS-resident sequence S_x (32 x 384 f32).
__device__ void mamba_layer(float* S_x, float* S_dt, float* S_dbl,
                            unsigned short* S_da, unsigned short* S_hn,
                            unsigned short* S_xz /* xi | z ; xi half reused as y */,
                            unsigned short* S_xc,
                            const LayerP& p, int T, int tid) {
  const int lane = tid & 31;
  const int wave = tid >> 5;

  // ---- Stage 0: RMSNorm -> bf16 A operand (rows >= T zero-padded) --------
  for (int t = wave; t < TMAX; t += 8) {
    if (t < T) {
      float ss = 0.f;
#pragma unroll
      for (int j = 0; j < DM / 32; ++j) {
        float v = S_x[t * DM + j * 32 + lane];
        ss += v * v;
      }
#pragma unroll
      for (int off = 16; off >= 1; off >>= 1) ss += __shfl_xor(ss, off, 32);
      float rinv = rsqrtf(ss * (1.f / DM) + 1e-5f);
#pragma unroll
      for (int j = 0; j < DM / 32; ++j) {
        int c = j * 32 + lane;
        S_hn[t * DM + c] = f2bf(S_x[t * DM + c] * p.norm_w[c] * rinv);
      }
    } else {
#pragma unroll
      for (int j = 0; j < DM / 32; ++j) S_hn[t * DM + j * 32 + lane] = 0;
    }
  }
  __syncthreads();

  // ---- Stage 1: xz = hn @ W_in   (32x384)x(384x1536) ---------------------
  for (int tile = wave; tile < (2 * DI) / 16; tile += 8) {   // 96 N-tiles
    v8f acc0 = {}; v8f acc1 = {};
#pragma unroll 1
    for (int kk = 0; kk < DM / 32; ++kk) {                   // 12 K-steps
      v16bf b  = load_b_pk(p.pW_in, 96, kk, tile, lane);
      v16bf a0 = load_a_lds(S_hn, DM, 0,  kk * 32, lane);
      v16bf a1 = load_a_lds(S_hn, DM, 16, kk * 32, lane);
      acc0 = wmma_bf16(a0, b, acc0);
      acc1 = wmma_bf16(a1, b, acc1);
    }
    int n = tile * 16 + (lane & 15);
#pragma unroll
    for (int v = 0; v < 8; ++v) {
      int m0 = ((lane >> 4) << 3) + v;
      S_xz[m0 * (2 * DI) + n]        = f2bf(acc0[v]);
      S_xz[(m0 + 16) * (2 * DI) + n] = f2bf(acc1[v]);
    }
  }
  __syncthreads();

  // ---- Stage 2: causal depthwise conv (k=4) + SiLU -> S_xc ---------------
  for (int idx = tid; idx < T * DI; idx += NTHREADS) {
    int t = idx / DI, c = idx % DI;
    float s = p.conv_b[c];
#pragma unroll
    for (int k = 0; k < 4; ++k) {
      int tt = t - 3 + k;
      if (tt >= 0) s += bf2f(S_xz[tt * (2 * DI) + c]) * p.conv_w[k * DI + c];
    }
    S_xc[t * DI + c] = f2bf(siluf(s));
  }
  __syncthreads();

  // ---- Stage 3: dbl = xc @ W_x   (32x768)x(768x56->64) -> S_dbl (32x64) --
  {
    int ntile = wave & 3, mt = wave >> 2;                    // 8 tiles, 8 waves
    int nbase = ntile * 16, mbase = mt * 16;
    v8f acc = {};
#pragma unroll 1
    for (int kk = 0; kk < DI / 32; ++kk) {
      v16bf a = load_a_lds(S_xc, DI, mbase, kk * 32, lane);
      v16bf b = load_b_pk(p.pW_x, 4, kk, ntile, lane);
      acc = wmma_bf16(a, b, acc);
    }
    int n = nbase + (lane & 15);
#pragma unroll
    for (int v = 0; v < 8; ++v) {
      int m = mbase + ((lane >> 4) << 3) + v;
      S_dbl[m * 64 + n] = acc[v];
    }
  }
  __syncthreads();

  // ---- Stage 4: pad dbl[:, :24] -> bf16 A (32x32) ------------------------
  for (int idx = tid; idx < TMAX * 32; idx += NTHREADS) {
    int t = idx >> 5, j = idx & 31;
    S_da[idx] = f2bf(j < DTR ? S_dbl[t * 64 + j] : 0.f);
  }
  __syncthreads();

  // ---- Stage 5: dt = softplus(dbl[:, :24] @ W_dt + b_dt) -----------------
  for (int tile = wave; tile < (DI / 16) * 2; tile += 8) {   // 96 tiles
    int ntile = tile >> 1, mt = tile & 1;
    int nbase = ntile * 16, mbase = mt * 16;
    v16bf a = load_a_lds(S_da, 32, mbase, 0, lane);
    v16bf b = load_b_pk(p.pW_dt, 48, 0, ntile, lane);
    v8f acc = {};
    acc = wmma_bf16(a, b, acc);
    int n = nbase + (lane & 15);
#pragma unroll
    for (int v = 0; v < 8; ++v) {
      int m = mbase + ((lane >> 4) << 3) + v;
      float d = acc[v] + p.b_dt[n];
      float sp = (d > 20.f) ? d : __logf(1.f + __expf(d));
      S_dt[m * DI + n] = (m < T) ? sp : 0.f;
    }
  }
  __syncthreads();

  // ---- Stage 6: selective scan (per-thread: 3 channels x 16 states) ------
  {
    int c0 = tid * (DI / NTHREADS);                          // 3 channels
    float Areg[3][DST], hst[3][DST];
#pragma unroll
    for (int i = 0; i < 3; ++i)
#pragma unroll
      for (int s = 0; s < DST; ++s) {
        Areg[i][s] = -__expf(p.A_log[(c0 + i) * DST + s]);
        hst[i][s]  = 0.f;
      }
    for (int t = 0; t < T; ++t) {
      float Bv[DST], Cv[DST];
#pragma unroll
      for (int s = 0; s < DST; ++s) {
        Bv[s] = S_dbl[t * 64 + DTR + s];
        Cv[s] = S_dbl[t * 64 + DTR + DST + s];
      }
#pragma unroll
      for (int i = 0; i < 3; ++i) {
        int c = c0 + i;
        float dtv = S_dt[t * DI + c];
        float xv  = bf2f(S_xc[t * DI + c]);
        float dx  = dtv * xv;
        float acc = 0.f;
#pragma unroll
        for (int s = 0; s < DST; ++s) {
          hst[i][s] = hst[i][s] * __expf(dtv * Areg[i][s]) + dx * Bv[s];
          acc += hst[i][s] * Cv[s];
        }
        S_dt[t * DI + c] = acc;    // ys overwrites dt in place
      }
    }
  }
  __syncthreads();

  // ---- Stage 7: y = (ys + xc*D) * silu(z) -> bf16 into xi half of S_xz ---
  for (int idx = tid; idx < T * DI; idx += NTHREADS) {
    int t = idx / DI, c = idx % DI;
    float y  = S_dt[idx] + bf2f(S_xc[idx]) * p.D[c];
    float zz = bf2f(S_xz[t * (2 * DI) + DI + c]);
    S_xz[t * (2 * DI) + c] = f2bf(y * siluf(zz));
  }
  __syncthreads();

  // ---- Stage 8: out = y @ W_out, residual add into S_x -------------------
  for (int tile = wave; tile < (DM / 16) * 2; tile += 8) {   // 48 tiles
    int ntile = tile >> 1, mt = tile & 1;
    int nbase = ntile * 16, mbase = mt * 16;
    v8f acc = {};
#pragma unroll 1
    for (int kk = 0; kk < DI / 32; ++kk) {
      v16bf a = load_a_lds(S_xz, 2 * DI, mbase, kk * 32, lane);
      v16bf b = load_b_pk(p.pW_out, 24, kk, ntile, lane);
      acc = wmma_bf16(a, b, acc);
    }
    int n = nbase + (lane & 15);
#pragma unroll
    for (int v = 0; v < 8; ++v) {
      int m = mbase + ((lane >> 4) << 3) + v;
      S_x[m * DM + n] += acc[v];
    }
  }
  __syncthreads();
}

// ---------------------------------------------------------------------------
// One block per fiber: gather -> fwd layer -> reverse -> bwd layer -> reverse
// -> scatter. Mask decides whether the fiber is processed.
// ---------------------------------------------------------------------------
__global__ __launch_bounds__(NTHREADS)
void mamba_axis_kernel(float* __restrict__ x, LayerP pf, LayerP pb,
                       const int* __restrict__ maskA, const int* __restrict__ maskB,
                       int dimA, int dimB, long strideN, long strideA,
                       long strideB, long strideT, int T) {
  int bid = blockIdx.x;
  int n   = bid / (dimA * dimB);
  int rem = bid % (dimA * dimB);
  int a   = rem / dimB;
  int b   = rem % dimB;
  if (!(maskA[a] | maskB[b])) return;

  extern __shared__ char smem[];
  float* S_x   = (float*)smem;                                   // 49152
  float* S_dt  = (float*)(smem + 49152);                         // 98304
  unsigned short* S_hn = (unsigned short*)(smem + 147456);       // 24576 region
  unsigned short* S_da = (unsigned short*)(smem + 147456);       //  (overlay)
  float* S_dbl = (float*)(smem + 147456 + 2048);                 //  (overlay)
  unsigned short* S_xz = (unsigned short*)(smem + 172032);       // 98304
  unsigned short* S_xc = (unsigned short*)(smem + 270336);       // 49152
                                                                 // = 319488 B
  const int tid = threadIdx.x;
  float* base = x + (size_t)n * strideN + (size_t)a * strideA + (size_t)b * strideB;

  // zero pad rows of S_xc once (conv only writes t < T)
  for (int idx = tid; idx < TMAX * DI; idx += NTHREADS) S_xc[idx] = 0;
  // gather fiber, zero-pad rows >= T
  for (int idx = tid; idx < TMAX * DM; idx += NTHREADS) {
    int t = idx / DM, d = idx % DM;
    S_x[idx] = (t < T) ? base[(size_t)t * strideT + d] : 0.f;
  }
  __syncthreads();

  mamba_layer(S_x, S_dt, S_dbl, S_da, S_hn, S_xz, S_xc, pf, T, tid);

  // reverse sequence
  for (int idx = tid; idx < (T / 2) * DM; idx += NTHREADS) {
    int t = idx / DM, d = idx % DM;
    float tmp = S_x[t * DM + d];
    S_x[t * DM + d] = S_x[(T - 1 - t) * DM + d];
    S_x[(T - 1 - t) * DM + d] = tmp;
  }
  __syncthreads();

  mamba_layer(S_x, S_dt, S_dbl, S_da, S_hn, S_xz, S_xc, pb, T, tid);

  for (int idx = tid; idx < (T / 2) * DM; idx += NTHREADS) {
    int t = idx / DM, d = idx % DM;
    float tmp = S_x[t * DM + d];
    S_x[t * DM + d] = S_x[(T - 1 - t) * DM + d];
    S_x[(T - 1 - t) * DM + d] = tmp;
  }
  __syncthreads();

  // scatter back
  for (int idx = tid; idx < T * DM; idx += NTHREADS) {
    int t = idx / DM, d = idx % DM;
    base[(size_t)t * strideT + d] = S_x[idx];
  }
}

// Pack one f32 row-major weight [kmax x nmax] (leading dim = nmax) into WMMA-B
// fragment order bf16: out[((kk*ntiles + nt)*32 + lane)*16 + j], where
// lane&15 = N-in-tile, lane>>4 = K-half, j = K-in-half. Pads K and N with 0.
__global__ void pack_b_kernel(const float* __restrict__ W,
                              unsigned short* __restrict__ out,
                              int kblocks, int ntiles, int kmax, int nmax) {
  int idx = blockIdx.x * blockDim.x + threadIdx.x;
  int total = kblocks * ntiles * 512;
  if (idx >= total) return;
  int j    = idx & 15;
  int lane = (idx >> 4) & 31;
  int nt   = (idx >> 9) % ntiles;
  int kk   = idx / (ntiles << 9);
  int nn   = nt * 16 + (lane & 15);
  int k    = kk * 32 + ((lane >> 4) << 4) + j;
  float f  = (k < kmax && nn < nmax) ? W[(size_t)k * nmax + nn] : 0.f;
  out[idx] = f2bf(f);
}

__global__ void build_masks_kernel(const int* __restrict__ li,
                                   const int* __restrict__ hi,
                                   const int* __restrict__ wi,
                                   int* lm, int* hm, int* wm) {
  int t = threadIdx.x;
  if (t < 32) lm[t] = 0;
  if (t < 28) { hm[t] = 0; wm[t] = 0; }
  __syncthreads();
  if (t < 4) lm[li[t] & 31] = 1;
  if (t < 3) { hm[hi[t] % 28] = 1; wm[wi[t] % 28] = 1; }
}

extern "C" void kernel_launch(void* const* d_in, const int* in_sizes, int n_in,
                              void* d_out, int out_size, void* d_ws, size_t ws_size,
                              hipStream_t stream) {
  (void)in_sizes; (void)n_in; (void)out_size; (void)ws_size;
  const int N = 4, L = 32, H = 28, W = 28;
  const long sW = DM, sH = (long)W * DM, sL = (long)H * W * DM, sN = (long)L * sL;

  // workspace layout: masks (1 KB) then packed bf16 weights per stack
  char* ws = (char*)d_ws;
  int* lm = (int*)ws; int* hm = lm + 64; int* wm = lm + 128;
  unsigned short* wp = (unsigned short*)(ws + 1024);

  // params flattened in dict order: x, then 6 stacks x 10 tensors, then indices
  // stack keys: norm_w, W_in, conv_w, conv_b, W_x, W_dt, b_dt, A_log, D, W_out
  LayerP P[6];
  const size_t SZ_IN = 12 * 96 * 512, SZ_X = 24 * 4 * 512,
               SZ_DT = 1 * 48 * 512,  SZ_OUT = 24 * 24 * 512;
  for (int s = 0; s < 6; ++s) {
    const float* const* q = (const float* const*)(d_in + 1 + 10 * s);
    P[s].norm_w = q[0]; P[s].conv_w = q[2]; P[s].conv_b = q[3];
    P[s].b_dt = q[6];   P[s].A_log = q[7];  P[s].D = q[8];
    P[s].pW_in  = wp;               wp += SZ_IN;
    P[s].pW_x   = wp;               wp += SZ_X;
    P[s].pW_dt  = wp;               wp += SZ_DT;
    P[s].pW_out = wp;               wp += SZ_OUT;
    // launch packers for this stack
    pack_b_kernel<<<(int)((SZ_IN  + 255) / 256), 256, 0, stream>>>(
        q[1], (unsigned short*)P[s].pW_in, 12, 96, 384, 1536);
    pack_b_kernel<<<(int)((SZ_X   + 255) / 256), 256, 0, stream>>>(
        q[4], (unsigned short*)P[s].pW_x, 24, 4, 768, 56);
    pack_b_kernel<<<(int)((SZ_DT  + 255) / 256), 256, 0, stream>>>(
        q[5], (unsigned short*)P[s].pW_dt, 1, 48, 24, 768);
    pack_b_kernel<<<(int)((SZ_OUT + 255) / 256), 256, 0, stream>>>(
        q[9], (unsigned short*)P[s].pW_out, 24, 24, 768, 384);
  }
  const int* l_idx = (const int*)d_in[61];
  const int* h_idx = (const int*)d_in[62];
  const int* w_idx = (const int*)d_in[63];
  build_masks_kernel<<<1, 64, 0, stream>>>(l_idx, h_idx, w_idx, lm, hm, wm);

  float* out = (float*)d_out;
  hipMemcpyAsync(out, d_in[0], (size_t)N * L * H * W * DM * sizeof(float),
                 hipMemcpyDeviceToDevice, stream);

  const size_t shmem = 319488;
  // axis L: fibers over (n, h, w), T = 32
  mamba_axis_kernel<<<N * H * W, NTHREADS, shmem, stream>>>(
      out, P[0], P[1], hm, wm, H, W, sN, sH, sW, sL, L);
  // axis H: fibers over (n, l, w), T = 28
  mamba_axis_kernel<<<N * L * W, NTHREADS, shmem, stream>>>(
      out, P[2], P[3], lm, wm, L, W, sN, sL, sW, sH, H);
  // axis W: fibers over (n, l, h), T = 28
  mamba_axis_kernel<<<N * L * H, NTHREADS, shmem, stream>>>(
      out, P[4], P[5], lm, hm, L, H, sN, sL, sH, sW, W);
}
